// MultiUserHierarchicalGRU_11690900980385
// MI455X (gfx1250) — matrix operand order repaired and checked by reading, not compiled, not run on hardware
//
#include <hip/hip_runtime.h>
#include <hip/hip_bf16.h>

typedef __attribute__((ext_vector_type(16))) _Float16 v16h;
typedef __attribute__((ext_vector_type(8)))  _Float16 v8h;
typedef __attribute__((ext_vector_type(8)))  float    v8f;

// ---------------------------------------------------------------------------
// Model dims
// ---------------------------------------------------------------------------
#define Bn   16
#define Rn   20
#define Sn   16
#define Wn   32
#define En   128
#define Hn   256
#define Gn   768         // 3*H
#define KREV 288         // H+20=276 padded to multiple of 32

// ---------------------------------------------------------------------------
// Embedding gather: tokens -> x f16 [ntok][128]
// ---------------------------------------------------------------------------
__global__ void embed_gather_kernel(const int* __restrict__ tokens,
                                    const float* __restrict__ emb,
                                    _Float16* __restrict__ x, int ntok) {
    int i = blockIdx.x * blockDim.x + threadIdx.x;
    int total = ntok * (En / 8);                 // 8 elements per thread-chunk
    int stride = gridDim.x * blockDim.x;
    for (; i < total; i += stride) {
        int t = i >> 4;                          // token index (128/8 = 16 chunks)
        int c = (i & 15) * 8;
        int tok = tokens[t];
        const float* s = emb + (size_t)tok * En + c;
        v8h o;
#pragma unroll
        for (int k = 0; k < 8; k++) o[k] = (_Float16)s[k];
        *(v8h*)(x + (size_t)t * En + c) = o;
    }
}

// ---------------------------------------------------------------------------
// f32 -> f16 weight convert with column zero-padding
// ---------------------------------------------------------------------------
__global__ void cvt_f16_pad_kernel(const float* __restrict__ src,
                                   _Float16* __restrict__ dst,
                                   int rows, int cin, int cout) {
    int i = blockIdx.x * blockDim.x + threadIdx.x;
    int total = rows * cout;
    int stride = gridDim.x * blockDim.x;
    for (; i < total; i += stride) {
        int r = i / cout, c = i - r * cout;
        dst[i] = (c < cin) ? (_Float16)src[(size_t)r * cin + c] : (_Float16)0.0f;
    }
}

// ---------------------------------------------------------------------------
// WMMA tile loaders (CDNA5 16-bit A 16x32 and B 32x16 layouts, wave32)
// ---------------------------------------------------------------------------
__device__ __forceinline__ v16h wmma_load_A(const _Float16* base, int ldk,
                                            int k0, int lane) {
    int row = lane & 15;
    int off = (lane < 16) ? 0 : 16;
    const _Float16* p = base + row * ldk + k0 + off;
    v8h lo = *(const v8h*)(p);        // K = k0+off .. +7
    v8h hi = *(const v8h*)(p + 8);    // K = k0+off+8 .. +15
    v16h a;
#pragma unroll
    for (int i = 0; i < 8; i++) { a[i] = lo[i]; a[8 + i] = hi[i]; }
    return a;
}

// B[k][n] = W[n][k] (W row-major [768][ldk]); lane covers N=lane&15,
// K = k0 + (lane<16?0:16) + e  -> 16 contiguous halves (32B aligned)
__device__ __forceinline__ v16h wmma_load_B(const _Float16* __restrict__ W,
                                            int ldk, int tile, int k0, int lane) {
    int n = tile * 16 + (lane & 15);
    int koff = k0 + ((lane < 16) ? 0 : 16);
    return *(const v16h*)(W + (size_t)n * ldk + koff);
}

__device__ __forceinline__ float fast_sigmoid(float x) {
    return 1.0f / (1.0f + __expf(-x));
}

// ---------------------------------------------------------------------------
// Persistent GRU kernel. One block = 16 batch rows, 8 waves, hidden state
// resident in LDS across all timesteps. x layout: [Mtot][T][Kin] f16 (padded).
// ---------------------------------------------------------------------------
__global__ __launch_bounds__(256)
void gru_wmma_kernel(const _Float16* __restrict__ x,
                     const _Float16* __restrict__ Wih,   // [768][Kin] f16
                     const _Float16* __restrict__ Whh,   // [768][256] f16
                     const float* __restrict__ bih,
                     const float* __restrict__ bhh,
                     float* __restrict__ h_out,          // [Mtot][256] f32
                     _Float16* __restrict__ h_out16,     // [Mtot][256] f16
                     int T, int Kin) {
    __shared__ __align__(16) _Float16 xs[16 * KREV];     // worst-case Kin=288
    __shared__ __align__(16) _Float16 hs16[16 * Hn];
    __shared__ __align__(16) float    hs32[16 * Hn];

    const int tid = threadIdx.x;
    const int lane = tid & 31;
    const int wave = tid >> 5;
    const int m0 = blockIdx.x * 16;
    const int kin_steps = Kin >> 5;
    const int hi = (lane >= 16) ? 1 : 0;

    // h0 = 0
    for (int i = tid; i < 16 * Hn; i += 256) {
        hs32[i] = 0.0f;
        hs16[i] = (_Float16)0.0f;
    }

    // wave w owns gate columns [32w, 32w+32): tiles r:{2w,2w+1} z:{16+..} n:{32+..}
    int tiles[6];
    tiles[0] = 2 * wave;      tiles[1] = 2 * wave + 1;
    tiles[2] = 16 + 2 * wave; tiles[3] = 16 + 2 * wave + 1;
    tiles[4] = 32 + 2 * wave; tiles[5] = 32 + 2 * wave + 1;

    float bi[6], bh[6];
#pragma unroll
    for (int j = 0; j < 6; j++) {
        int c = tiles[j] * 16 + (lane & 15);
        bi[j] = bih[c];
        bh[j] = bhh[c];
    }

    for (int t = 0; t < T; t++) {
        __syncthreads();
        // stage x_t tile [16][Kin] into LDS (Kin multiple of 8 -> v8h chunks)
        {
            int chunks = (16 * Kin) >> 3;
            for (int i = tid; i < chunks; i += 256) {
                int idx = i << 3;
                int row = idx / Kin;
                int col = idx - row * Kin;
                const v8h* src =
                    (const v8h*)(x + ((size_t)(m0 + row) * T + t) * Kin + col);
                *(v8h*)(xs + row * Kin + col) = *src;
            }
        }
        __syncthreads();

        v8f gi[6], gh[6];
#pragma unroll
        for (int j = 0; j < 6; j++) {
            v8f vi, vh;
#pragma unroll
            for (int d = 0; d < 8; d++) { vi[d] = bi[j]; vh[d] = bh[j]; }
            gi[j] = vi;
            gh[j] = vh;
        }

        // gi = x_t @ Wih^T + bih
        for (int kk = 0; kk < kin_steps; kk++) {
            v16h a = wmma_load_A(xs, Kin, kk * 32, lane);
#pragma unroll
            for (int j = 0; j < 6; j++) {
                v16h b = wmma_load_B(Wih, Kin, tiles[j], kk * 32, lane);
                gi[j] = __builtin_amdgcn_wmma_f32_16x16x32_f16(
                    false, a, false, b, (short)0, gi[j], false, false);
            }
        }
        // gh = h @ Whh^T + bhh   (K = 256 -> 8 steps)
        for (int kk = 0; kk < 8; kk++) {
            v16h a = wmma_load_A(hs16, Hn, kk * 32, lane);
#pragma unroll
            for (int j = 0; j < 6; j++) {
                v16h b = wmma_load_B(Whh, Hn, tiles[j], kk * 32, lane);
                gh[j] = __builtin_amdgcn_wmma_f32_16x16x32_f16(
                    false, a, false, b, (short)0, gh[j], false, false);
            }
        }
        __syncthreads();   // all waves done reading h before we overwrite it

        // gates + hidden update, fully in-register per wave
#pragma unroll
        for (int p = 0; p < 2; p++) {
            int j = 32 * wave + 16 * p + (lane & 15);
#pragma unroll
            for (int d = 0; d < 8; d++) {
                int m = d + 8 * hi;
                float r = fast_sigmoid(gi[p][d] + gh[p][d]);
                float z = fast_sigmoid(gi[2 + p][d] + gh[2 + p][d]);
                float n = tanhf(gi[4 + p][d] + r * gh[4 + p][d]);
                float ho = hs32[m * Hn + j];
                float hn = (1.0f - z) * n + z * ho;
                hs32[m * Hn + j] = hn;
                hs16[m * Hn + j] = (_Float16)hn;
            }
        }
    }

    __syncthreads();
    for (int i = tid; i < 16 * Hn; i += 256) {
        int row = i >> 8, col = i & 255;
        h_out[(size_t)(m0 + row) * Hn + col] = hs32[i];
        h_out16[(size_t)(m0 + row) * Hn + col] = hs16[i];
    }
}

// ---------------------------------------------------------------------------
// Build p = [h_rev | normalize(user_feats)], emit p f16 (padded to 288),
// and compute r_stars = selu(p@w1^T + b1)@w2^T + b2. One block per (b,r).
// ---------------------------------------------------------------------------
__global__ __launch_bounds__(128)
void p_rfc_kernel(const float* __restrict__ h_rev,   // [320][256]
                  const float* __restrict__ uf,      // [16][20][20]
                  const float* __restrict__ w1,      // [128][276]
                  const float* __restrict__ b1,      // [128]
                  const float* __restrict__ w2,      // [128]
                  const float* __restrict__ b2,      // [1]
                  _Float16* __restrict__ p16,        // [320][288]
                  float* __restrict__ r_out) {       // [320]
    __shared__ float prow[KREV];
    __shared__ float red[128];
    const int q = blockIdx.x;
    const int tid = threadIdx.x;

    for (int c = tid; c < Hn; c += 128) prow[c] = h_rev[(size_t)q * Hn + c];
    if (tid == 0) {
        float ss = 0.0f;
        for (int k = 0; k < 20; k++) { float v = uf[q * 20 + k]; ss += v * v; }
        red[0] = fmaxf(sqrtf(ss), 1e-12f);
    }
    __syncthreads();
    if (tid < 20) prow[Hn + tid] = uf[q * 20 + tid] / red[0];
    if (tid >= 20 && tid < 32) prow[Hn + tid] = 0.0f;   // pad 276..287
    __syncthreads();

    for (int c = tid; c < KREV; c += 128)
        p16[(size_t)q * KREV + c] = (_Float16)prow[c];

    float acc = b1[tid];
    for (int k = 0; k < 276; k++) acc += prow[k] * w1[tid * 276 + k];
    const float kScale = 1.0507009873554805f, kAlpha = 1.6732632423543772f;
    float s = (acc > 0.0f) ? kScale * acc : kScale * kAlpha * (__expf(acc) - 1.0f);
    red[tid] = s * w2[tid];
    __syncthreads();
    for (int st = 64; st > 0; st >>= 1) {
        if (tid < st) red[tid] += red[tid + st];
        __syncthreads();
    }
    if (tid == 0) r_out[q] = red[0] + b2[0];
}

// ---------------------------------------------------------------------------
// b_stars = selu(h_biz @ p_w1^T + p_b1) @ p_w2^T + p_b2. One block per b.
// ---------------------------------------------------------------------------
__global__ __launch_bounds__(128)
void bfc_kernel(const float* __restrict__ h_biz,     // [16][256]
                const float* __restrict__ w1,        // [128][256]
                const float* __restrict__ b1,        // [128]
                const float* __restrict__ w2,        // [128]
                const float* __restrict__ b2,        // [1]
                float* __restrict__ out) {           // [16]
    __shared__ float red[128];
    const int b = blockIdx.x;
    const int tid = threadIdx.x;
    float acc = b1[tid];
    for (int k = 0; k < Hn; k++) acc += h_biz[(size_t)b * Hn + k] * w1[tid * Hn + k];
    const float kScale = 1.0507009873554805f, kAlpha = 1.6732632423543772f;
    float s = (acc > 0.0f) ? kScale * acc : kScale * kAlpha * (__expf(acc) - 1.0f);
    red[tid] = s * w2[tid];
    __syncthreads();
    for (int st = 64; st > 0; st >>= 1) {
        if (tid < st) red[tid] += red[tid + st];
        __syncthreads();
    }
    if (tid == 0) out[b] = red[0] + b2[0];
}

// ---------------------------------------------------------------------------
// Launch
// ---------------------------------------------------------------------------
extern "C" void kernel_launch(void* const* d_in, const int* in_sizes, int n_in,
                              void* d_out, int out_size, void* d_ws, size_t ws_size,
                              hipStream_t stream) {
    const int*   inputs = (const int*)d_in[0];
    const float* uf     = (const float*)d_in[1];
    const float* emb    = (const float*)d_in[2];
    const float* wWih   = (const float*)d_in[3];
    const float* wWhh   = (const float*)d_in[4];
    const float* wbih   = (const float*)d_in[5];
    const float* wbhh   = (const float*)d_in[6];
    const float* sWih   = (const float*)d_in[7];
    const float* sWhh   = (const float*)d_in[8];
    const float* sbih   = (const float*)d_in[9];
    const float* sbhh   = (const float*)d_in[10];
    const float* rWih   = (const float*)d_in[11];
    const float* rWhh   = (const float*)d_in[12];
    const float* rbih   = (const float*)d_in[13];
    const float* rbhh   = (const float*)d_in[14];
    const float* pw1    = (const float*)d_in[15];
    const float* pb1    = (const float*)d_in[16];
    const float* pw2    = (const float*)d_in[17];
    const float* pb2    = (const float*)d_in[18];
    const float* rfw1   = (const float*)d_in[19];
    const float* rfb1   = (const float*)d_in[20];
    const float* rfw2   = (const float*)d_in[21];
    const float* rfb2   = (const float*)d_in[22];

    float* b_stars = (float*)d_out;          // [16]
    float* r_stars = (float*)d_out + Bn;     // [320]

    // workspace carve-out (256B aligned)
    char* ws = (char*)d_ws;
    size_t off = 0;
    auto carve = [&](size_t bytes) {
        void* p = ws + off;
        off += (bytes + 255) & ~(size_t)255;
        return p;
    };
    const size_t NTOK = (size_t)Bn * Rn * Sn * Wn;       // 163840
    const size_t NSEQ = (size_t)Bn * Rn * Sn;            // 5120
    const size_t NREV = (size_t)Bn * Rn;                 // 320

    _Float16* x16      = (_Float16*)carve(NTOK * En * 2);
    _Float16* wWih16   = (_Float16*)carve((size_t)Gn * En * 2);
    _Float16* wWhh16   = (_Float16*)carve((size_t)Gn * Hn * 2);
    _Float16* sWih16   = (_Float16*)carve((size_t)Gn * Hn * 2);
    _Float16* sWhh16   = (_Float16*)carve((size_t)Gn * Hn * 2);
    _Float16* rWih16   = (_Float16*)carve((size_t)Gn * KREV * 2);
    _Float16* rWhh16   = (_Float16*)carve((size_t)Gn * Hn * 2);
    float*    h_sent   = (float*)   carve(NSEQ * Hn * 4);
    _Float16* h_sent16 = (_Float16*)carve(NSEQ * Hn * 2);
    float*    h_rev    = (float*)   carve(NREV * Hn * 4);
    _Float16* h_rev16  = (_Float16*)carve(NREV * Hn * 2);
    _Float16* p16      = (_Float16*)carve(NREV * KREV * 2);
    float*    h_biz    = (float*)   carve((size_t)Bn * Hn * 4);
    _Float16* h_biz16  = (_Float16*)carve((size_t)Bn * Hn * 2);

    // 1. embedding gather -> f16 x
    embed_gather_kernel<<<4096, 256, 0, stream>>>(inputs, emb, x16, (int)NTOK);

    // 2. weight conversion (f32 -> f16, K-padded where needed)
    cvt_f16_pad_kernel<<<512, 256, 0, stream>>>(wWih, wWih16, Gn, En, En);
    cvt_f16_pad_kernel<<<512, 256, 0, stream>>>(wWhh, wWhh16, Gn, Hn, Hn);
    cvt_f16_pad_kernel<<<512, 256, 0, stream>>>(sWih, sWih16, Gn, Hn, Hn);
    cvt_f16_pad_kernel<<<512, 256, 0, stream>>>(sWhh, sWhh16, Gn, Hn, Hn);
    cvt_f16_pad_kernel<<<512, 256, 0, stream>>>(rWih, rWih16, Gn, Hn + 20, KREV);
    cvt_f16_pad_kernel<<<512, 256, 0, stream>>>(rWhh, rWhh16, Gn, Hn, Hn);

    // 3. word GRU: 5120 sequences, T=32, Kin=128
    gru_wmma_kernel<<<(int)(NSEQ / 16), 256, 0, stream>>>(
        x16, wWih16, wWhh16, wbih, wbhh, h_sent, h_sent16, Wn, En);

    // 4. sentence GRU: 320 reviews, T=16, Kin=256 (consumes h_sent16 as x)
    gru_wmma_kernel<<<(int)(NREV / 16), 256, 0, stream>>>(
        h_sent16, sWih16, sWhh16, sbih, sbhh, h_rev, h_rev16, Sn, Hn);

    // 5. build p (+normalize user_feats) + per-review FC head
    p_rfc_kernel<<<(int)NREV, 128, 0, stream>>>(
        h_rev, uf, rfw1, rfb1, rfw2, rfb2, p16, r_stars);

    // 6. review GRU: batch 16, T=20, Kin=288 (padded)
    gru_wmma_kernel<<<1, 256, 0, stream>>>(
        p16, rWih16, rWhh16, rbih, rbhh, h_biz, h_biz16, Rn, KREV);

    // 7. business FC head
    bfc_kernel<<<Bn, 128, 0, stream>>>(h_biz, pw1, pb1, pw2, pb2, b_stars);
}
